// TransformerBlock_47588237640426
// MI455X (gfx1250) — compile-verified, matrix operand order checked by reading, not run on hardware
//
#include <hip/hip_runtime.h>
#include <math.h>

// ---------------------------------------------------------------------------
// Transformer block for MI455X (gfx1250, wave32, WMMA).
// All matmuls use v_wmma_f32_16x16x32_bf16 (f32 accumulate).
// Attention is flash-style (no T x T materialization); softmax row-sums are
// computed on the matrix pipe (P x ones WMMA) instead of lane shuffles.
// ---------------------------------------------------------------------------

typedef __attribute__((ext_vector_type(16))) __bf16 v16bf;
typedef __attribute__((ext_vector_type(8)))  float  v8f;
typedef __attribute__((ext_vector_type(4)))  unsigned int uint4v;
typedef __attribute__((ext_vector_type(4)))  float  v4f;

union Frag {
    v16bf  bf;
    uint4v u[2];
};

__device__ __forceinline__ unsigned short f32_to_bf16(float f) {
    union { float f; unsigned int u; } v;
    v.f = f;
    unsigned int u = v.u;
    u += 0x7FFFu + ((u >> 16) & 1u);   // round-to-nearest-even
    return (unsigned short)(u >> 16);
}

// A/B fragment loader for v_wmma_f32_16x16x32_bf16.
// 16-bit A-matrix 16x32 layout (ISA 7.12.2):
//   lanes 0-15 : row = lane,    K = {0..7} then {16..23}
//   lanes 16-31: row = lane-16, K = {8..15} then {24..31}
// => two 16-byte contiguous loads per lane from a row-major [16 x >=32] tile.
__device__ __forceinline__ Frag load_frag(const unsigned short* base, int ld, int lane) {
    int half8 = ((lane >> 4) & 1) * 8;
    int r     = lane & 15;
    const unsigned short* p = base + (size_t)r * ld + half8;
    Frag f;
    f.u[0] = *(const uint4v*)(p);        // K 0..7   (or 8..15)
    f.u[1] = *(const uint4v*)(p + 16);   // K 16..23 (or 24..31)
    return f;
}

__device__ __forceinline__ v8f wmma_bf16(const Frag& a, const Frag& b, v8f c) {
    return __builtin_amdgcn_wmma_f32_16x16x32_bf16(
        /*neg_a=*/false, a.bf, /*neg_b=*/false, b.bf,
        /*c_mod=*/(short)0, c, /*reuse_a=*/false, /*reuse_b=*/false);
}

// ---------------------------------------------------------------------------
// fp32 (g, R, C) -> bf16 (g, C, R)   (weight transpose+downconvert)
// ---------------------------------------------------------------------------
__global__ __launch_bounds__(256)
void convert_transpose_kernel(const float* __restrict__ in,
                              unsigned short* __restrict__ out,
                              int R, int C) {
    size_t slab = (size_t)R * C;
    const float* ip = in + (size_t)blockIdx.y * slab;
    unsigned short* op = out + (size_t)blockIdx.y * slab;
    int idx = blockIdx.x * 256 + threadIdx.x;
    if (idx < R * C) {
        int r = idx / C;
        int c = idx - r * C;
        op[(size_t)c * R + r] = f32_to_bf16(ip[idx]);
    }
}

// ---------------------------------------------------------------------------
// LayerNorm: fp32 row (1024) -> bf16 row.  One block (8 waves) per row.
// ---------------------------------------------------------------------------
__global__ __launch_bounds__(256)
void layernorm_bf16_kernel(const float* __restrict__ x,
                           const float* __restrict__ gam,
                           const float* __restrict__ bet,
                           unsigned short* __restrict__ out) {
    int row = blockIdx.x;
    const float* xr = x + (size_t)row * 1024;
    int c0 = threadIdx.x * 4;
    v4f v = *(const v4f*)(xr + c0);
    float s  = v[0] + v[1] + v[2] + v[3];
    float s2 = v[0]*v[0] + v[1]*v[1] + v[2]*v[2] + v[3]*v[3];
    #pragma unroll
    for (int m = 16; m >= 1; m >>= 1) {
        s  += __shfl_xor(s,  m, 32);
        s2 += __shfl_xor(s2, m, 32);
    }
    __shared__ float red[2][8];
    int wave = threadIdx.x >> 5, lane = threadIdx.x & 31;
    if (lane == 0) { red[0][wave] = s; red[1][wave] = s2; }
    __syncthreads();
    float ts = 0.f, ts2 = 0.f;
    #pragma unroll
    for (int i = 0; i < 8; i++) { ts += red[0][i]; ts2 += red[1][i]; }
    float mu   = ts * (1.0f / 1024.0f);
    float var  = ts2 * (1.0f / 1024.0f) - mu * mu;
    float rstd = rsqrtf(var + 1e-5f);
    #pragma unroll
    for (int i = 0; i < 4; i++) {
        float h = (v[i] - mu) * rstd * gam[c0 + i] + bet[c0 + i];
        out[(size_t)row * 1024 + c0 + i] = f32_to_bf16(h);
    }
}

// ---------------------------------------------------------------------------
// GEMM: C[M,N] = A[M,K] @ Bt[N,K]^T   (both bf16, N-major weights, f32 accum)
// Block 256 thr = 8 waves; tile 128(M) x 128(N); K staged 64 at a time in
// 32 KB LDS.  Software pipelined: next stage's global loads are issued into
// registers between the barriers, overlapping the current stage's 16 WMMAs.
// Wave tile 32(M) x 64(N) = 8 accumulators.
// MODE 0: bf16 out, +bias, optional ReLU           (FFN up-proj)
// MODE 1: f32 out = acc + bias + residual          (Wo proj, FFN down-proj)
// MODE 2: bf16 scatter to (B,H,T,HS)               (Q, K projections)
// MODE 3: bf16 scatter to (B,H,HS,T)               (V projection, transposed)
// ---------------------------------------------------------------------------
template<int MODE, bool RELU>
__global__ __launch_bounds__(256)
void gemm_bf16_kernel(const unsigned short* __restrict__ A,
                      const unsigned short* __restrict__ Bt,
                      const float* __restrict__ bias,
                      const float* __restrict__ residual,
                      unsigned short* __restrict__ outB,
                      float* __restrict__ outF,
                      int M, int N, int K) {
    __shared__ unsigned short sA[128 * 64];   // 16 KB
    __shared__ unsigned short sB[128 * 64];   // 16 KB

    int tid  = threadIdx.x;
    int lane = tid & 31;
    int wave = tid >> 5;
    int wm   = wave >> 1;     // 0..3  -> M offset wm*32
    int wn   = wave & 1;      // 0..1  -> N offset wn*64
    int m0   = blockIdx.y * 128;
    int n0   = blockIdx.x * 128;

    v8f acc[2][4] = {};

    // Each thread stages 32 elements of A and of B per K-stage (4 b128 each).
    int ldRow = tid >> 1;            // 0..127
    int ldCol = (tid & 1) * 32;      // 0 or 32

    const unsigned short* aG = A  + (size_t)(m0 + ldRow) * K + ldCol;
    const unsigned short* bG = Bt + (size_t)(n0 + ldRow) * K + ldCol;
    unsigned short* sAp = sA + ldRow * 64 + ldCol;
    unsigned short* sBp = sB + ldRow * 64 + ldCol;

    uint4v ra[4], rb[4];
    #pragma unroll
    for (int i = 0; i < 4; i++) {
        ra[i] = *(const uint4v*)(aG + i * 8);
        rb[i] = *(const uint4v*)(bG + i * 8);
    }

    int nStages = K >> 6;
    for (int s = 0; s < nStages; s++) {
        __syncthreads();               // previous stage fully consumed
        #pragma unroll
        for (int i = 0; i < 4; i++) {
            *(uint4v*)(sAp + i * 8) = ra[i];
            *(uint4v*)(sBp + i * 8) = rb[i];
        }
        if (s + 1 < nStages) {         // pipeline next stage into registers
            const unsigned short* aN = aG + (size_t)(s + 1) * 64;
            const unsigned short* bN = bG + (size_t)(s + 1) * 64;
            #pragma unroll
            for (int i = 0; i < 4; i++) {
                ra[i] = *(const uint4v*)(aN + i * 8);
                rb[i] = *(const uint4v*)(bN + i * 8);
            }
        }
        __syncthreads();               // stage visible to all waves

        #pragma unroll
        for (int kk = 0; kk < 2; kk++) {
            Frag aF0 = load_frag(sA + (wm * 32 +  0) * 64 + kk * 32, 64, lane);
            Frag aF1 = load_frag(sA + (wm * 32 + 16) * 64 + kk * 32, 64, lane);
            #pragma unroll
            for (int ni = 0; ni < 4; ni++) {
                Frag bF = load_frag(sB + (wn * 64 + ni * 16) * 64 + kk * 32, 64, lane);
                acc[0][ni] = wmma_bf16(aF0, bF, acc[0][ni]);
                acc[1][ni] = wmma_bf16(aF1, bF, acc[1][ni]);
            }
        }
    }

    // Epilogue.  C layout: lanes 0-15 -> M=r, lanes 16-31 -> M=8+r; N=lane&15.
    int l15   = lane & 15;
    int mHalf = ((lane >> 4) & 1) * 8;
    #pragma unroll
    for (int mi = 0; mi < 2; mi++) {
        #pragma unroll
        for (int ni = 0; ni < 4; ni++) {
            int col = n0 + wn * 64 + ni * 16 + l15;
            float bval = 0.0f;
            if constexpr (MODE == 0 || MODE == 1) bval = bias[col];
            #pragma unroll
            for (int r = 0; r < 8; r++) {
                int row = m0 + wm * 32 + mi * 16 + mHalf + r;
                float v = acc[mi][ni][r];
                if constexpr (MODE == 0) {
                    v += bval;
                    if (RELU) v = fmaxf(v, 0.0f);
                    outB[(size_t)row * N + col] = f32_to_bf16(v);
                } else if constexpr (MODE == 1) {
                    outF[(size_t)row * N + col] =
                        v + bval + residual[(size_t)row * N + col];
                } else {
                    int b = row >> 11, t = row & 2047;     // T = 2048
                    int h = col >> 6,  hs = col & 63;      // HS = 64
                    size_t idx = (MODE == 2)
                        ? ((size_t)((b * 16 + h) * 2048 + t)) * 64 + hs
                        : ((size_t)((b * 16 + h) * 64 + hs)) * 2048 + t;
                    outB[idx] = f32_to_bf16(v);
                }
            }
        }
    }
}

// ---------------------------------------------------------------------------
// Flash attention (causal).  One wave per 16 query rows; keys streamed 32 at
// a time.  Q/K in (B,H,T,HS) bf16; V pre-transposed (B,H,HS,T) bf16.
// Row-sum of exp(P) computed via WMMA against an all-ones matrix.
// Output: bf16 (B,T,H*HS) = (BT x 1024), heads concatenated.
// ---------------------------------------------------------------------------
__global__ __launch_bounds__(256)
void attention_kernel(const unsigned short* __restrict__ q,
                      const unsigned short* __restrict__ k,
                      const unsigned short* __restrict__ vt,
                      unsigned short* __restrict__ out) {
    __shared__ unsigned short sP[8][16 * 32];   // per-wave P staging (1 KB each)

    int wave = threadIdx.x >> 5;
    int lane = threadIdx.x & 31;
    int bh   = blockIdx.y;                       // b*16 + h
    int q0   = (blockIdx.x * 8 + wave) * 16;

    const unsigned short* qp = q  + ((size_t)bh * 2048 + q0) * 64;
    const unsigned short* kp = k  + (size_t)bh * 2048 * 64;
    const unsigned short* vp = vt + (size_t)bh * 64 * 2048;

    Frag qf0 = load_frag(qp,      64, lane);     // K = hs 0..31
    Frag qf1 = load_frag(qp + 32, 64, lane);     // K = hs 32..63

    Frag ones;                                   // bf16 1.0 broadcast (B = ones)
    ones.u[0] = (uint4v){0x3F803F80u, 0x3F803F80u, 0x3F803F80u, 0x3F803F80u};
    ones.u[1] = ones.u[0];

    float mrow[8], lrow[8];
    #pragma unroll
    for (int r = 0; r < 8; r++) { mrow[r] = -__builtin_inff(); lrow[r] = 0.0f; }
    v8f o[4] = {};

    const float scale = 0.125f;                  // HS^-0.5
    int nIdx  = lane & 15;
    int mHalf = ((lane >> 4) & 1) * 8;
    unsigned short* sp = sP[wave];

    for (int s0 = 0; s0 <= q0; s0 += 32) {
        __builtin_prefetch(kp + (size_t)(s0 + 32) * 64, 0, 1);  // next key tile

        // S = Q @ K^T for keys [s0, s0+31]; two 16-key groups.
        v8f sAcc = {}, sBcc = {};
        Frag kf;
        kf = load_frag(kp + (size_t)(s0)      * 64,      64, lane);
        sAcc = wmma_bf16(qf0, kf, sAcc);
        kf = load_frag(kp + (size_t)(s0)      * 64 + 32, 64, lane);
        sAcc = wmma_bf16(qf1, kf, sAcc);
        kf = load_frag(kp + (size_t)(s0 + 16) * 64,      64, lane);
        sBcc = wmma_bf16(qf0, kf, sBcc);
        kf = load_frag(kp + (size_t)(s0 + 16) * 64 + 32, 64, lane);
        sBcc = wmma_bf16(qf1, kf, sBcc);

        // Online softmax (row = mHalf+r lives across 16 lanes of a half-wave).
        float alphaA[8], p0[8], p1[8];
        #pragma unroll
        for (int r = 0; r < 8; r++) {
            int   qq = q0 + mHalf + r;
            float a  = sAcc[r] * scale;
            float b  = sBcc[r] * scale;
            if (s0 + nIdx      > qq) a = -__builtin_inff();
            if (s0 + 16 + nIdx > qq) b = -__builtin_inff();
            float rm = fmaxf(a, b);
            rm = fmaxf(rm, __shfl_xor(rm, 1, 32));
            rm = fmaxf(rm, __shfl_xor(rm, 2, 32));
            rm = fmaxf(rm, __shfl_xor(rm, 4, 32));
            rm = fmaxf(rm, __shfl_xor(rm, 8, 32));
            float mnew = fmaxf(mrow[r], rm);
            alphaA[r]  = __expf(mrow[r] - mnew);
            mrow[r]    = mnew;
            p0[r] = __expf(a - mnew);
            p1[r] = __expf(b - mnew);
            #pragma unroll
            for (int j = 0; j < 4; j++) o[j][r] *= alphaA[r];
        }

        // C-layout -> A-fragment layout via per-wave LDS round trip.
        #pragma unroll
        for (int r = 0; r < 8; r++) {
            sp[(mHalf + r) * 32 + nIdx]      = f32_to_bf16(p0[r]);
            sp[(mHalf + r) * 32 + 16 + nIdx] = f32_to_bf16(p1[r]);
        }
        asm volatile("s_wait_dscnt 0" ::: "memory");  // per-wave DS ordering
        Frag pf = load_frag(sp, 32, lane);

        // Row sums of P on the matrix pipe: rsum[r] = sum_key P[row][key].
        v8f rsum = {};
        rsum = wmma_bf16(pf, ones, rsum);
        #pragma unroll
        for (int r = 0; r < 8; r++) lrow[r] = lrow[r] * alphaA[r] + rsum[r];

        // O += P @ V  (V^T rows are hs, K contiguous over keys).
        #pragma unroll
        for (int j = 0; j < 4; j++) {
            Frag vf = load_frag(vp + (size_t)(j * 16) * 2048 + s0, 2048, lane);
            o[j] = wmma_bf16(pf, vf, o[j]);
        }
    }

    // Normalize and scatter to (B,T,H*HS).
    int b = bh >> 4, h = bh & 15;
    #pragma unroll
    for (int j = 0; j < 4; j++) {
        #pragma unroll
        for (int r = 0; r < 8; r++) {
            int t  = q0 + mHalf + r;
            int hs = j * 16 + nIdx;
            out[((size_t)(b * 2048 + t)) * 1024 + h * 64 + hs] =
                f32_to_bf16(o[j][r] * (1.0f / lrow[r]));
        }
    }
}

// ---------------------------------------------------------------------------
// Host-side orchestration.
// ---------------------------------------------------------------------------
extern "C" void kernel_launch(void* const* d_in, const int* in_sizes, int n_in,
                              void* d_out, int out_size, void* d_ws, size_t ws_size,
                              hipStream_t stream) {
    (void)in_sizes; (void)n_in; (void)out_size; (void)ws_size;
    const float* x   = (const float*)d_in[0];
    const float* wq  = (const float*)d_in[1];
    const float* wk  = (const float*)d_in[2];
    const float* wv  = (const float*)d_in[3];
    const float* wo  = (const float*)d_in[4];
    const float* bo  = (const float*)d_in[5];
    const float* w1  = (const float*)d_in[6];
    const float* b1  = (const float*)d_in[7];
    const float* w2  = (const float*)d_in[8];
    const float* b2  = (const float*)d_in[9];
    const float* g1p = (const float*)d_in[10];
    const float* be1 = (const float*)d_in[11];
    const float* g2p = (const float*)d_in[12];
    const float* be2 = (const float*)d_in[13];
    float* out = (float*)d_out;

    // Workspace layout (152 MB total; ff1 reuses the q/k/vt/attn region).
    char* ws = (char*)d_ws;
    const size_t MB = 1ull << 20;
    unsigned short* wqt  = (unsigned short*)(ws +   0 * MB);  //  2 MB
    unsigned short* wkt  = (unsigned short*)(ws +   2 * MB);  //  2 MB
    unsigned short* wvt  = (unsigned short*)(ws +   4 * MB);  //  2 MB
    unsigned short* wot  = (unsigned short*)(ws +   6 * MB);  //  2 MB
    unsigned short* w1t  = (unsigned short*)(ws +   8 * MB);  //  8 MB
    unsigned short* w2t  = (unsigned short*)(ws +  16 * MB);  //  8 MB
    unsigned short* hb   = (unsigned short*)(ws +  24 * MB);  // 16 MB  LN1(x) bf16
    unsigned short* qb   = (unsigned short*)(ws +  40 * MB);  // 16 MB
    unsigned short* kbuf = (unsigned short*)(ws +  56 * MB);  // 16 MB
    unsigned short* vtb  = (unsigned short*)(ws +  72 * MB);  // 16 MB
    unsigned short* attn = (unsigned short*)(ws +  88 * MB);  // 16 MB
    float*          x1   = (float*)         (ws + 104 * MB);  // 32 MB
    unsigned short* h2   = (unsigned short*)(ws + 136 * MB);  // 16 MB
    unsigned short* ff1  = (unsigned short*)(ws +  40 * MB);  // 64 MB (reuse)

    dim3 blk(256);

    // Weights: fp32 -> bf16, transposed to N-major (K contiguous).
    convert_transpose_kernel<<<dim3((1024 * 64 + 255) / 256, 16), blk, 0, stream>>>(wq, wqt, 1024, 64);
    convert_transpose_kernel<<<dim3((1024 * 64 + 255) / 256, 16), blk, 0, stream>>>(wk, wkt, 1024, 64);
    convert_transpose_kernel<<<dim3((1024 * 64 + 255) / 256, 16), blk, 0, stream>>>(wv, wvt, 1024, 64);
    convert_transpose_kernel<<<dim3((1024 * 1024) / 256, 1), blk, 0, stream>>>(wo, wot, 1024, 1024);
    convert_transpose_kernel<<<dim3((1024 * 4096) / 256, 1), blk, 0, stream>>>(w1, w1t, 1024, 4096);
    convert_transpose_kernel<<<dim3((4096 * 1024) / 256, 1), blk, 0, stream>>>(w2, w2t, 4096, 1024);

    // h = LN1(x)  (bf16)
    layernorm_bf16_kernel<<<8192, blk, 0, stream>>>(x, g1p, be1, hb);

    // Q, K, V projections (M=8192, N=1024, K=1024); block tile 128x128.
    dim3 gP(8, 64);
    gemm_bf16_kernel<2, false><<<gP, blk, 0, stream>>>(hb, wqt, nullptr, nullptr, qb,   nullptr, 8192, 1024, 1024);
    gemm_bf16_kernel<2, false><<<gP, blk, 0, stream>>>(hb, wkt, nullptr, nullptr, kbuf, nullptr, 8192, 1024, 1024);
    gemm_bf16_kernel<3, false><<<gP, blk, 0, stream>>>(hb, wvt, nullptr, nullptr, vtb,  nullptr, 8192, 1024, 1024);

    // Causal flash attention.
    attention_kernel<<<dim3(16, 64), blk, 0, stream>>>(qb, kbuf, vtb, attn);

    // x1 = x + attn @ wo + bo   (f32)
    gemm_bf16_kernel<1, false><<<gP, blk, 0, stream>>>(attn, wot, bo, x, nullptr, x1, 8192, 1024, 1024);

    // h2 = LN2(x1)  (bf16)
    layernorm_bf16_kernel<<<8192, blk, 0, stream>>>(x1, g2p, be2, h2);

    // ff1 = relu(h2 @ w1 + b1)  (bf16, M=8192, N=4096, K=1024)
    gemm_bf16_kernel<0, true><<<dim3(32, 64), blk, 0, stream>>>(h2, w1t, b1, nullptr, ff1, nullptr, 8192, 4096, 1024);

    // out = x1 + ff1 @ w2 + b2  (f32, K=4096)
    gemm_bf16_kernel<1, false><<<gP, blk, 0, stream>>>(ff1, w2t, b2, x1, nullptr, out, 8192, 1024, 4096);
}
